// LSTMModel_68169720923005
// MI455X (gfx1250) — matrix operand order repaired and checked by reading, not compiled, not run on hardware
//
#include <hip/hip_runtime.h>
#include <math.h>

typedef __attribute__((ext_vector_type(16))) __bf16 v16bf;
typedef __attribute__((ext_vector_type(8)))  __bf16 v8bf;
typedef __attribute__((ext_vector_type(8)))  float  v8f;

#define L_SEQ 65536
#define KCAT  32
#define IDIM  256
#define HDIM  512
#define ODIM  256
#define KIN   288   // KCAT + IDIM
#define G4H   2048  // 4*HDIM

#if defined(__gfx1250__) && __has_builtin(__builtin_amdgcn_global_load_async_to_lds_b32) && __has_builtin(__builtin_amdgcn_s_wait_asynccnt)
#define HAVE_ASYNC_LDS 1
typedef __attribute__((address_space(1))) int* gptr_t;
typedef __attribute__((address_space(3))) int* lptr_t;
#else
#define HAVE_ASYNC_LDS 0
#endif

// ---------------- prep kernels ----------------

__global__ void prep_wcomb(const float* __restrict__ W_ih, const float* __restrict__ W_ctx,
                           __bf16* __restrict__ Wcomb) {
  int idx = blockIdx.x * blockDim.x + threadIdx.x;
  if (idx >= G4H * KIN) return;
  int n = idx / KIN, k = idx % KIN;
  float s = 0.f;
  for (int j = 0; j < IDIM; ++j) s = fmaf(W_ih[n * IDIM + j], W_ctx[j * KIN + k], s);
  Wcomb[idx] = (__bf16)s;
}

__global__ void prep_bcomb(const float* __restrict__ W_ih, const float* __restrict__ b_ctx,
                           const float* __restrict__ b_ih, float* __restrict__ bcomb) {
  int n = blockIdx.x * blockDim.x + threadIdx.x;
  if (n >= G4H) return;
  float s = b_ih[n];
  for (int j = 0; j < IDIM; ++j) s = fmaf(W_ih[n * IDIM + j], b_ctx[j], s);
  bcomb[n] = s;
}

__global__ void cvt_bf16(const float* __restrict__ src, __bf16* __restrict__ dst, int n) {
  int i = blockIdx.x * blockDim.x + threadIdx.x;
  if (i < n) dst[i] = (__bf16)src[i];
}

__global__ void init_counter(unsigned* c) { *c = 0u; }

// ---------------- gx GEMM: [L,288]bf16 x [288,2048]bf16 -> gx bf16 ----------------
// block = 256 threads (8 waves); each wave computes a 16x128 tile (8 WMMA accs);
// grid (L/16, 2).

__global__ __launch_bounds__(256) void gemm_gx(
    const float* __restrict__ cat, const float* __restrict__ inp,
    const __bf16* __restrict__ Wcomb, const float* __restrict__ bcomb,
    __bf16* __restrict__ gx) {
  const int lane  = threadIdx.x & 31;
  const int wave  = threadIdx.x >> 5;
  const int lr    = lane & 15;
  const int hi    = lane >> 4;                 // 0 or 1
  const int m     = blockIdx.x * 16 + lr;      // A row served by this lane
  const int nBase = blockIdx.y * 1024 + wave * 128;

  v8f acc[8];
  #pragma unroll
  for (int nt = 0; nt < 8; ++nt) {
    float b = bcomb[nBase + nt * 16 + lr];
    #pragma unroll
    for (int e = 0; e < 8; ++e) acc[nt][e] = b;
  }

  for (int ks = 0; ks < 9; ++ks) {
    const int kk  = ks * 32;
    const int khi = hi * 16;
    const float *aptr0, *aptr1;
    if (ks == 0) {                       // k-range 0..31 lives entirely in `category`
      aptr0 = cat + (size_t)m * KCAT + khi;
      aptr1 = cat + (size_t)m * KCAT + 8 + khi;
    } else {                             // k-range 32..287 lives in `input`
      const int kb = kk - KCAT;
      aptr0 = inp + (size_t)m * IDIM + kb + khi;
      aptr1 = inp + (size_t)m * IDIM + kb + 8 + khi;
    }
    float4 a00 = *(const float4*)aptr0;
    float4 a01 = *(const float4*)(aptr0 + 4);
    float4 a10 = *(const float4*)aptr1;
    float4 a11 = *(const float4*)(aptr1 + 4);
    v16bf af;
    af[0] = (__bf16)a00.x; af[1]  = (__bf16)a00.y; af[2]  = (__bf16)a00.z; af[3]  = (__bf16)a00.w;
    af[4] = (__bf16)a01.x; af[5]  = (__bf16)a01.y; af[6]  = (__bf16)a01.z; af[7]  = (__bf16)a01.w;
    af[8] = (__bf16)a10.x; af[9]  = (__bf16)a10.y; af[10] = (__bf16)a10.z; af[11] = (__bf16)a10.w;
    af[12]= (__bf16)a11.x; af[13] = (__bf16)a11.y; af[14] = (__bf16)a11.z; af[15] = (__bf16)a11.w;

    const int kb2 = kk + hi * 16;
    const __bf16* bbase = Wcomb + (size_t)(nBase + lr) * KIN + kb2;
    #pragma unroll
    for (int nt = 0; nt < 8; ++nt) {
      v16bf bf = *(const v16bf*)(bbase + (size_t)nt * 16 * KIN);
      acc[nt] = __builtin_amdgcn_wmma_f32_16x16x32_bf16(
          false, af, false, bf, (short)0, acc[nt], false, false);
    }
  }

  #pragma unroll
  for (int nt = 0; nt < 8; ++nt) {
    const int col = nBase + nt * 16 + lr;
    #pragma unroll
    for (int e = 0; e < 8; ++e) {
      const int row = blockIdx.x * 16 + e + hi * 8;
      gx[(size_t)row * G4H + col] = (__bf16)acc[nt][e];
    }
  }
}

// ---------------- persistent LSTM scan: 16 WGs x 1024 threads ----------------
// WG w owns h-slice [w*32, w*32+32); its 128 gate rows of W_hh live in LDS (bf16).
// gx gate inputs are double-buffered in LDS via async global->LDS loads (ASYNCcnt).

__global__ __launch_bounds__(1024, 1) void lstm_scan(
    const float* __restrict__ Whh, const float* __restrict__ bhh,
    const float* __restrict__ hidden0, const float* __restrict__ cell0,
    const __bf16* __restrict__ gx, __bf16* __restrict__ hs,
    __bf16* __restrict__ hbuf, unsigned* __restrict__ counter,
    float* __restrict__ hn, float* __restrict__ cn) {
  __shared__ __bf16 sW[128 * 516];   // pitch 516 -> dword stride 258 (conflict-free)
  __shared__ __bf16 sH[512];
  __shared__ __bf16 sGx[2][128];     // double-buffered gate inputs for this WG
  __shared__ float  sPart[8 * 128];
  __shared__ float  sGate[128];
  __shared__ float  sBhh[128];

  const int tid   = threadIdx.x;
  const int w     = blockIdx.x;      // 0..15
  const int hbase = w * 32;

  for (int idx = tid; idx < 128 * 512; idx += 1024) {
    int r = idx >> 9, k = idx & 511;
    int grow = (r >> 5) * HDIM + hbase + (r & 31);
    sW[r * 516 + k] = (__bf16)Whh[(size_t)grow * HDIM + k];
  }
  // thread j < 64 owns gate rows (2j, 2j+1): contiguous bf16 pair in gx.
  int prow = 0;      // global gx column of row 2j (pair base)
  if (tid < 64) {
    int r0 = tid * 2;
    prow = (r0 >> 5) * HDIM + hbase + (r0 & 31);
  }
  if (tid < 128) sBhh[tid] = bhh[(tid >> 5) * HDIM + hbase + (tid & 31)];
  if (tid < 512) sH[tid] = (__bf16)hidden0[tid];
  float c = 0.f;
  if (tid < 32) c = cell0[hbase + tid];

#if HAVE_ASYNC_LDS
  if (tid < 64) {
    __builtin_amdgcn_global_load_async_to_lds_b32(
        (gptr_t)(gx + prow), (lptr_t)&sGx[0][tid * 2], 0, 0);
  }
#endif
  __syncthreads();

  const int q = tid >> 7;            // k-chunk 0..7 (wave-uniform)
  const int r = tid & 127;           // gate row within slice
  const unsigned* wp  = (const unsigned*)(sW + r * 516 + q * 64);
  const unsigned* hp0 = (const unsigned*)sH + q * 32;

  for (int t = 0; t < L_SEQ; ++t) {
    float acc = 0.f;
    #pragma unroll 8
    for (int i = 0; i < 32; ++i) {
      unsigned wb = wp[i];
      unsigned hb = hp0[i];
      float w0 = __uint_as_float(wb << 16);
      float w1 = __uint_as_float(wb & 0xffff0000u);
      float h0 = __uint_as_float(hb << 16);
      float h1 = __uint_as_float(hb & 0xffff0000u);
      acc = fmaf(w0, h0, acc);
      acc = fmaf(w1, h1, acc);
    }
    sPart[q * 128 + r] = acc;
    __syncthreads();

    if (tid < 64) {
      const int r0 = tid * 2, r1 = r0 + 1;
      float g0 = sBhh[r0];
      float g1 = sBhh[r1];
      #pragma unroll
      for (int qq = 0; qq < 8; ++qq) {
        g0 += sPart[qq * 128 + r0];
        g1 += sPart[qq * 128 + r1];
      }
#if HAVE_ASYNC_LDS
      __builtin_amdgcn_s_wait_asynccnt(0);   // per-wave: same lanes issued the load
      g0 += (float)sGx[t & 1][r0];
      g1 += (float)sGx[t & 1][r1];
      if (t + 1 < L_SEQ) {
        __builtin_amdgcn_global_load_async_to_lds_b32(
            (gptr_t)(gx + (size_t)(t + 1) * G4H + prow),
            (lptr_t)&sGx[(t + 1) & 1][r0], 0, 0);
      }
#else
      g0 += (float)gx[(size_t)t * G4H + prow];
      g1 += (float)gx[(size_t)t * G4H + prow + 1];
      if (t + 1 < L_SEQ) __builtin_prefetch(&gx[(size_t)(t + 1) * G4H + prow], 0, 0);
#endif
      sGate[r0] = g0;
      sGate[r1] = g1;
    }
    __syncthreads();

    if (tid < 32) {
      float ig = 1.f / (1.f + __expf(-sGate[tid]));
      float fg = 1.f / (1.f + __expf(-sGate[32 + tid]));
      float gg = tanhf(sGate[64 + tid]);
      float og = 1.f / (1.f + __expf(-sGate[96 + tid]));
      c = fmaf(fg, c, ig * gg);
      float h = og * tanhf(c);
      __bf16 hb = (__bf16)h;
      hs[(size_t)t * HDIM + hbase + tid] = hb;
      hbuf[(t & 1) * HDIM + hbase + tid] = hb;
      if (t == L_SEQ - 1) { hn[hbase + tid] = h; cn[hbase + tid] = c; }
      __threadfence();
    }
    __syncthreads();

    if (t + 1 < L_SEQ) {
      if (tid == 0) {
        __hip_atomic_fetch_add(counter, 1u, __ATOMIC_RELEASE, __HIP_MEMORY_SCOPE_AGENT);
        unsigned target = 16u * (unsigned)(t + 1);
        while (__hip_atomic_load(counter, __ATOMIC_ACQUIRE, __HIP_MEMORY_SCOPE_AGENT) < target)
          __builtin_amdgcn_s_sleep(1);
      }
      __syncthreads();
      if (tid < 512) sH[tid] = hbuf[(t & 1) * HDIM + tid];
      __syncthreads();
    }
  }
}

// ---------------- FC + log_softmax: 16 rows per block, WMMA ----------------

__global__ __launch_bounds__(256) void fc_logsoftmax(
    const __bf16* __restrict__ hs, const __bf16* __restrict__ Wfc,
    const float* __restrict__ bfc, float* __restrict__ out) {
  __shared__ float sL[16][ODIM + 9];
  __shared__ float sRed[256];
  __shared__ float sMax[16];
  __shared__ float sSum[16];

  const int lane  = threadIdx.x & 31;
  const int wave  = threadIdx.x >> 5;
  const int lr    = lane & 15;
  const int hi    = lane >> 4;
  const int mTile = blockIdx.x * 16;
  const int m     = mTile + lr;

  v8f acc[2];
  #pragma unroll
  for (int nt = 0; nt < 2; ++nt)
    #pragma unroll
    for (int e = 0; e < 8; ++e) acc[nt][e] = 0.f;

  for (int ks = 0; ks < 16; ++ks) {
    const int kk  = ks * 32;
    const int khi = hi * 16;
    v8bf lo  = *(const v8bf*)(hs + (size_t)m * HDIM + kk + khi);
    v8bf hi8 = *(const v8bf*)(hs + (size_t)m * HDIM + kk + 8 + khi);
    v16bf af = __builtin_shufflevector(lo, hi8, 0,1,2,3,4,5,6,7,8,9,10,11,12,13,14,15);
    const int kb2 = kk + hi * 16;
    #pragma unroll
    for (int nt = 0; nt < 2; ++nt) {
      const int col = wave * 32 + nt * 16 + lr;
      v16bf bf = *(const v16bf*)(Wfc + (size_t)col * HDIM + kb2);
      acc[nt] = __builtin_amdgcn_wmma_f32_16x16x32_bf16(
          false, af, false, bf, (short)0, acc[nt], false, false);
    }
  }
  #pragma unroll
  for (int nt = 0; nt < 2; ++nt) {
    const int col = wave * 32 + nt * 16 + lr;
    float b = bfc[col];
    #pragma unroll
    for (int e = 0; e < 8; ++e) sL[e + hi * 8][col] = acc[nt][e] + b;
  }
  __syncthreads();

  const int row = threadIdx.x >> 4;
  const int sub = threadIdx.x & 15;
  float lmax = -3.4e38f;
  for (int i = 0; i < 16; ++i) lmax = fmaxf(lmax, sL[row][sub * 16 + i]);
  sRed[threadIdx.x] = lmax;
  __syncthreads();
  if (sub == 0) {
    float mx = sRed[row * 16];
    for (int i = 1; i < 16; ++i) mx = fmaxf(mx, sRed[row * 16 + i]);
    sMax[row] = mx;
  }
  __syncthreads();
  float mx = sMax[row];
  float lsum = 0.f;
  for (int i = 0; i < 16; ++i) lsum += __expf(sL[row][sub * 16 + i] - mx);
  sRed[threadIdx.x] = lsum;
  __syncthreads();
  if (sub == 0) {
    float s = 0.f;
    for (int i = 0; i < 16; ++i) s += sRed[row * 16 + i];
    sSum[row] = __logf(s);
  }
  __syncthreads();
  const float lse = sSum[row];
  for (int i = 0; i < 16; ++i) {
    int colx = sub * 16 + i;
    out[(size_t)(mTile + row) * ODIM + colx] = sL[row][colx] - mx - lse;
  }
}

// ---------------- launcher ----------------

extern "C" void kernel_launch(void* const* d_in, const int* in_sizes, int n_in,
                              void* d_out, int out_size, void* d_ws, size_t ws_size,
                              hipStream_t stream) {
  (void)in_sizes; (void)n_in; (void)out_size; (void)ws_size;
  const float* cat  = (const float*)d_in[0];
  const float* inp  = (const float*)d_in[1];
  const float* hid0 = (const float*)d_in[2];
  const float* cel0 = (const float*)d_in[3];
  const float* Wctx = (const float*)d_in[4];
  const float* bctx = (const float*)d_in[5];
  const float* Wih  = (const float*)d_in[6];
  const float* Whh  = (const float*)d_in[7];
  const float* bih  = (const float*)d_in[8];
  const float* bhh  = (const float*)d_in[9];
  const float* Wfc  = (const float*)d_in[10];
  const float* bfc  = (const float*)d_in[11];

  char* ws = (char*)d_ws;
  __bf16*   Wcomb = (__bf16*)(ws + 0);                 // 2048*288*2  = 1,179,648
  float*    bcomb = (float*)(ws + 1179648);            // 2048*4      = 8,192
  __bf16*   Wfcb  = (__bf16*)(ws + 1187840);           // 256*512*2   = 262,144
  __bf16*   gx    = (__bf16*)(ws + 1449984);           // 65536*2048*2= 268,435,456
  __bf16*   hs    = (__bf16*)(ws + 269885440);         // 65536*512*2 = 67,108,864
  __bf16*   hbuf  = (__bf16*)(ws + 336994304);         // 2*512*2     = 2,048
  unsigned* cnt   = (unsigned*)(ws + 336996352);       // 4

  float* out = (float*)d_out;
  float* hn  = out + (size_t)L_SEQ * ODIM;
  float* cn  = hn + HDIM;

  prep_wcomb<<<(G4H * KIN + 255) / 256, 256, 0, stream>>>(Wih, Wctx, Wcomb);
  prep_bcomb<<<(G4H + 255) / 256, 256, 0, stream>>>(Wih, bctx, bih, bcomb);
  cvt_bf16<<<(ODIM * HDIM + 255) / 256, 256, 0, stream>>>(Wfc, Wfcb, ODIM * HDIM);
  init_counter<<<1, 1, 0, stream>>>(cnt);

  dim3 g1(L_SEQ / 16, 2);
  gemm_gx<<<g1, 256, 0, stream>>>(cat, inp, Wcomb, bcomb, gx);
  lstm_scan<<<16, 1024, 0, stream>>>(Whh, bhh, hid0, cel0, gx, hs, hbuf, cnt, hn, cn);
  fc_logsoftmax<<<L_SEQ / 16, 256, 0, stream>>>(hs, Wfcb, bfc, out);
}
